// similarity_visualize_uncertainty_71193377898742
// MI455X (gfx1250) — compile-verified, matrix operand order checked by reading, not run on hardware
//
#include <hip/hip_runtime.h>
#include <math.h>

// ---------------------------------------------------------------------------
// Problem constants (grd/sat: [8, 32, 256, 256] f32, SHIFT=8 -> 17x17 shifts)
// ---------------------------------------------------------------------------
#define NSAMP   8
#define CCH     32
#define HK      256
#define WK      256
#define SHFT    8
#define NSH     17            // shifts per dimension
#define NS      289           // 17*17 total shifts
#define SGRP    19            // ceil(289/16) shift groups (WMMA tile rows = 16 shifts)
#define KSPLIT  4             // channel split across blocks
#define CPB     (CCH/KSPLIT)  // channels per block
#define RING_W  272           // 256 + 2*8 zero-padded row in LDS
#define RING_S  18            // ring slots (17 live rows + 1 staging row)
#define PER_SAMPLE ((size_t)CCH*HK*WK)   // 2,097,152

typedef __bf16 v16bf __attribute__((ext_vector_type(16)));
typedef float  v8f   __attribute__((ext_vector_type(8)));
typedef unsigned int u32x4 __attribute__((ext_vector_type(4)));
typedef int          i32x4 __attribute__((ext_vector_type(4)));
typedef int          i32x8 __attribute__((ext_vector_type(8)));

// ---------------------------------------------------------------------------
// Tensor Data Mover: DMA one 256-element f32 row (global -> LDS).
// D# group0: count=1 | lds_addr | global_addr | type=2 ("image")
// D# group1: data_size=4B, tensor_dim0=256, tensor_dim1=1, tile_dim0=256,
//            tile_dim1=1, tensor_dim0_stride=256.
// ---------------------------------------------------------------------------
__device__ __forceinline__ void tdm_load_row(unsigned lds_off, unsigned long long ga) {
    u32x4 g0;
    g0[0] = 1u;                                      // count=1 (valid), user mode
    g0[1] = lds_off;                                 // lds_addr [63:32]
    g0[2] = (unsigned)(ga & 0xffffffffull);          // global_addr [95:64]
    g0[3] = (unsigned)((ga >> 32) & 0x01ffffffull)   // global_addr [120:96]
          | (2u << 30);                              // type=2 [127:126]
    i32x8 g1;
    g1[0] = 0x00020000;          // workgroup_mask=0, data_size=2 (4 bytes)
    g1[1] = (int)(256u << 16);   // tensor_dim0 = 256  (bits 79:48)
    g1[2] = (int)(1u << 16);     // tensor_dim1 = 1    (bits 111:80)
    g1[3] = (int)(256u << 16);   // tile_dim0  = 256   (bits 127:112)
    g1[4] = 1;                   // tile_dim1  = 1     (bits 143:128)
    g1[5] = 256;                 // tensor_dim0_stride = 256 (bits 207:160)
    g1[6] = 0;
    g1[7] = 0;
    i32x4 gz4 = {0, 0, 0, 0};
#if __has_include(<hip/amd_detail/amd_gfx1250_TDM.h>)
    i32x8 gz8 = {0, 0, 0, 0, 0, 0, 0, 0};
    __builtin_amdgcn_tensor_load_to_lds(g0, g1, gz4, gz4, gz8, 0);
#else
    __builtin_amdgcn_tensor_load_to_lds(g0, g1, gz4, gz4, 0);
#endif
}

// ---------------------------------------------------------------------------
// Zero the correlation accumulator (ws is poisoned by the harness)
// ---------------------------------------------------------------------------
__global__ void zero_f32(float* __restrict__ p, int n) {
    int i = blockIdx.x * 256 + threadIdx.x;
    if (i < n) p[i] = 0.f;
}

// ---------------------------------------------------------------------------
// Per-sample L2 norms of grd (sat norm cancels analytically)
// ---------------------------------------------------------------------------
__global__ __launch_bounds__(256) void grd_norms_k(const float* __restrict__ grd,
                                                   float* __restrict__ norms) {
    const int m = blockIdx.x, t = threadIdx.x;
    const float* p = grd + (size_t)m * PER_SAMPLE;
    float acc = 0.f;
    for (size_t k = t; k < PER_SAMPLE; k += 256) { float v = p[k]; acc += v * v; }
    __shared__ float red[256];
    red[t] = acc; __syncthreads();
    for (int off = 128; off > 0; off >>= 1) {
        if (t < off) red[t] += red[t + off];
        __syncthreads();
    }
    if (t == 0) norms[m] = fmaxf(sqrtf(red[0]), 1e-12f);
}

// ---------------------------------------------------------------------------
// Convert grd f32 -> bf16, pre-swizzled into the exact WMMA B-operand register
// order: chunk (c,i,jb) holds 32 lanes x 16 contiguous bf16 (32B per lane).
// Lane l: column n = l%16 (n>=8 zero-padded), half = l/16, element e -> K(e,half)
// with K = (e<8 ? e : e+8) + 8*half per the CDNA5 16-bit fragment layout.
// ---------------------------------------------------------------------------
__global__ __launch_bounds__(256) void grd_swizzle_bf16(const float* __restrict__ grd,
                                                        __bf16* __restrict__ gbf) {
    const int ci = blockIdx.x;              // c*256 + i  (8192 blocks)
    const int t  = threadIdx.x;
    const int jb = t >> 5, lane = t & 31;
    const int n = lane & 15, half = lane >> 4;
    const int c = ci >> 8, i = ci & 255;
    const size_t base = ((size_t)ci * 8 + jb) * 512 + (size_t)lane * 16;
#pragma unroll
    for (int e = 0; e < 16; ++e) {
        const int j = jb * 32 + ((e < 8) ? e : e + 8) + 8 * half;
        float v = 0.f;
        if (n < 8) v = grd[(((size_t)n * CCH + c) * HK + i) * WK + j];
        gbf[base + e] = (__bf16)v;
    }
}

// ---------------------------------------------------------------------------
// Per-pixel channel energy of sat, then 2D inclusive prefix sums so the
// sliding-window energy sums become O(1) lookups.
// ---------------------------------------------------------------------------
__global__ __launch_bounds__(256) void sat_energy_k(const float* __restrict__ sat,
                                                    float* __restrict__ E) {
    const int m = blockIdx.x, i = blockIdx.y, j = threadIdx.x;
    float acc = 0.f;
    for (int c = 0; c < CCH; ++c) {
        float v = sat[(((size_t)m * CCH + c) * HK + i) * WK + j];
        acc += v * v;
    }
    E[((size_t)m * HK + i) * WK + j] = acc;
}

__global__ __launch_bounds__(256) void integral2d_k(float* __restrict__ E) {
    const int m = blockIdx.x, t = threadIdx.x;
    float* S = E + (size_t)m * HK * WK;
    float run = 0.f;
    for (int j = 0; j < WK; ++j) { run += S[(size_t)t * WK + j]; S[(size_t)t * WK + j] = run; }
    __syncthreads();
    run = 0.f;
    for (int i = 0; i < HK; ++i) { run += S[(size_t)i * WK + t]; S[(size_t)i * WK + t] = run; }
}

// ---------------------------------------------------------------------------
// Main kernel: all-pairs shifted cross-correlation via v_wmma_f32_16x16x32_bf16.
//   D[16x16] : rows = 16 shifts of this group, cols = 8 grd samples (8 padded)
//   K        : (c, i, j) contracted in chunks of 32 along j
// Block (m, shift-group, c-split), 8 waves; wave w owns j-block w.
// An 18-slot rolling f32 halo of sat[m,c] rows lives in LDS (+-8 col zero pad);
// wave 0 stages one new row per i step with the Tensor Data Mover
// (tensor_load_to_lds + s_wait_tensorcnt), staged one slot ahead of the 17
// live rows so a single workgroup barrier per step suffices.
// ---------------------------------------------------------------------------
__global__ __launch_bounds__(256) void corr_wmma_k(const float* __restrict__ sat,
                                                   const __bf16* __restrict__ gbf,
                                                   float* __restrict__ corr) {
    __shared__ float ring[RING_S * RING_W];   // 18 * 272 * 4 = 19,584 B

    const int m  = blockIdx.x;
    const int sg = blockIdx.y;
    const int cz = blockIdx.z;
    const int t  = threadIdx.x;
    const int wave = t >> 5, lane = t & 31;
    const int Arow = lane & 15;           // A-matrix row held by this lane
    const int half = lane >> 4;
    const int s = sg * 16 + Arow;         // shift index for this lane's A row
    const bool valid = (s < NS);
    const int dy = valid ? (s / NSH - SHFT) : 0;
    const int dx = valid ? (s % NSH - SHFT) : 0;
    const int jb = wave;
    const int colbase = jb * 32 + dx + SHFT + 8 * half;  // LDS column base

    // Zero the whole ring once: the +-8 column pads stay zero forever, and
    // slots 0..7 start as the (zero) rows -8..-1.
    for (int p = t; p < RING_S * RING_W; p += 256) ring[p] = 0.f;
    __syncthreads();

    v8f acc = {0.f, 0.f, 0.f, 0.f, 0.f, 0.f, 0.f, 0.f};

    for (int cc = 0; cc < CPB; ++cc) {
        const int c = cz * CPB + cc;
        const float* satbase = sat + ((size_t)m * CCH + c) * HK * WK;

        // Prefill: rows -8..-1 -> zero data columns of slots 0..7 (stale from
        // previous channel); rows 0..8 -> TDM into slots 8..16.
        if (cc != 0) {
            for (int idx = t; idx < 8 * 256; idx += 256)
                ring[(idx >> 8) * RING_W + 8 + (idx & 255)] = 0.f;
        }
        if (wave == 0) {
            for (int r = 0; r <= 8; ++r) {
                unsigned lds_off = (unsigned)(size_t)(const void*)&ring[(r + 8) * RING_W + 8];
                unsigned long long ga =
                    (unsigned long long)(size_t)(const void*)(satbase + (size_t)r * WK);
                tdm_load_row(lds_off, ga);
            }
            __builtin_amdgcn_s_wait_tensorcnt(0);
        }
        __syncthreads();

        for (int i = 0; i < HK; ++i) {
            // ---- compute step i: rows i-8..i+8 live in slots (i..i+16)%18 ----
            v16bf a;
            if (valid) {
                const int slot = (i + dy + 8) % RING_S;
                const float* rp = &ring[slot * RING_W + colbase];
#pragma unroll
                for (int e = 0; e < 16; ++e) a[e] = (__bf16)rp[(e < 8) ? e : e + 8];
            } else {
#pragma unroll
                for (int e = 0; e < 16; ++e) a[e] = (__bf16)0.f;
            }

            const v16bf b = *(const v16bf*)(gbf +
                ((((size_t)c * HK + i) * 8 + jb) * 512 + (size_t)lane * 16));

            acc = __builtin_amdgcn_wmma_f32_16x16x32_bf16(
                false, a, false, b, (short)0, acc, false, false);

            // ---- stage row i+9 into slot (i+17)%18 (disjoint from read set) ----
            {
                const int r = i + 9;
                const int slot = (i + 17) % RING_S;
                if (wave == 0) {
                    if (r < HK) {
                        unsigned lds_off =
                            (unsigned)(size_t)(const void*)&ring[slot * RING_W + 8];
                        unsigned long long ga =
                            (unsigned long long)(size_t)(const void*)(satbase + (size_t)r * WK);
                        tdm_load_row(lds_off, ga);
                        __builtin_amdgcn_s_wait_tensorcnt(0);
                    } else {
                        for (int q = lane; q < 256; q += 32)
                            ring[slot * RING_W + 8 + q] = 0.f;
                    }
                }
            }
            __syncthreads();
        }
    }

    // Writeback: D layout — lane l, VGPR v -> row M = v + 8*(l>=16), col N = l%16
    const int N = lane & 15;
    if (N < 8) {
#pragma unroll
        for (int v = 0; v < 8; ++v) {
            const int ss = sg * 16 + v + 8 * half;
            if (ss < NS)
                atomicAdd(&corr[((size_t)m * NS + ss) * 8 + N], acc[v]);
        }
    }
}

// ---------------------------------------------------------------------------
// Finalize: normalize by grd norm * sqrt(window energy), max over shifts per
// (m,n), diagonal argmax peak -> (rows, cols).
// ---------------------------------------------------------------------------
__global__ __launch_bounds__(512) void finalize_k(const float* __restrict__ corr,
                                                  const float* __restrict__ S,
                                                  const float* __restrict__ norms,
                                                  float* __restrict__ out_sim,
                                                  int* __restrict__ peaks) {
    const int m = blockIdx.x, t = threadIdx.x;
    const float* Sm = S + (size_t)m * HK * WK;
    const bool on = (t < NS);
    float vals[8];
    if (on) {
        const int dy = t / NSH - SHFT, dx = t % NSH - SHFT;
        const int r1 = dy > 0 ? dy : 0, r2 = (HK - 1 + dy) < (HK - 1) ? (HK - 1 + dy) : (HK - 1);
        const int c1 = dx > 0 ? dx : 0, c2 = (WK - 1 + dx) < (WK - 1) ? (WK - 1 + dx) : (WK - 1);
        auto Sv = [&](int r, int c) -> float { return (r < 0 || c < 0) ? 0.f : Sm[(size_t)r * WK + c]; };
        const float win = Sv(r2, c2) - Sv(r1 - 1, c2) - Sv(r2, c1 - 1) + Sv(r1 - 1, c1 - 1);
        const float denom = fmaxf(sqrtf(fmaxf(win, 0.f)), 1e-12f);
        for (int n = 0; n < 8; ++n)
            vals[n] = corr[((size_t)m * NS + t) * 8 + n] / (norms[n] * denom);
    } else {
        for (int n = 0; n < 8; ++n) vals[n] = -3.4e38f;
    }

    __shared__ float red[512];
    for (int n = 0; n < 8; ++n) {
        red[t] = vals[n]; __syncthreads();
        for (int off = 256; off > 0; off >>= 1) {
            if (t < off) red[t] = fmaxf(red[t], red[t + off]);
            __syncthreads();
        }
        if (t == 0) out_sim[m * 8 + n] = red[0];
        __syncthreads();
    }

    __shared__ float dv[512];
    __shared__ int   di[512];
    dv[t] = on ? vals[m] : -3.4e38f;
    di[t] = t;
    __syncthreads();
    for (int off = 256; off > 0; off >>= 1) {
        if (t < off) {
            const float a = dv[t], b = dv[t + off];
            const int ia = di[t], ib = di[t + off];
            if (b > a || (b == a && ib < ia)) { dv[t] = b; di[t] = ib; }
        }
        __syncthreads();
    }
    if (t == 0) { peaks[2 * m] = di[0] / NSH; peaks[2 * m + 1] = di[0] % NSH; }
}

// ---------------------------------------------------------------------------
// Mask scatter: out[m,a,b] = grd_mask[m, a+8-rows, b+8-cols] (0 outside).
// grd_mask = |sum_c grd_raw| > 1e-6 * norm_grd[m].
// ---------------------------------------------------------------------------
__global__ __launch_bounds__(256) void make_mask_k(const float* __restrict__ grd,
                                                   const float* __restrict__ norms,
                                                   const int* __restrict__ peaks,
                                                   float* __restrict__ outmask) {
    const int m = blockIdx.x, a = blockIdx.y, b = threadIdx.x;
    const int i = a + SHFT - peaks[2 * m];
    const int j = b + SHFT - peaks[2 * m + 1];
    float val = 0.f;
    if (i >= 0 && i < HK && j >= 0 && j < WK) {
        float ssum = 0.f;
        for (int c = 0; c < CCH; ++c)
            ssum += grd[(((size_t)m * CCH + c) * HK + i) * WK + j];
        val = (fabsf(ssum) > 1e-6f * norms[m]) ? 1.f : 0.f;
    }
    outmask[((size_t)m * HK + a) * WK + b] = val;
}

// ---------------------------------------------------------------------------
// Launch
// ---------------------------------------------------------------------------
extern "C" void kernel_launch(void* const* d_in, const int* in_sizes, int n_in,
                              void* d_out, int out_size, void* d_ws, size_t ws_size,
                              hipStream_t stream) {
    (void)in_sizes; (void)n_in; (void)out_size; (void)ws_size;
    const float* grd = (const float*)d_in[0];
    const float* sat = (const float*)d_in[1];
    float* out = (float*)d_out;                 // [64 sim] ++ [8*256*256 mask]

    char* ws = (char*)d_ws;
    const size_t GBF_BYTES  = (size_t)CCH * HK * 8 * 512 * 2;   // 67,108,864
    const size_t CORR_BYTES = (size_t)NSAMP * NS * 8 * 4;       // 73,984 (256-aligned)
    const size_t E_BYTES    = (size_t)NSAMP * HK * WK * 4;      // 2,097,152
    __bf16* gbf  = (__bf16*)(ws);
    float*  corr = (float*)(ws + GBF_BYTES);
    float*  E    = (float*)(ws + GBF_BYTES + CORR_BYTES);
    float*  norms= (float*)(ws + GBF_BYTES + CORR_BYTES + E_BYTES);
    int*    peaks= (int*)  (ws + GBF_BYTES + CORR_BYTES + E_BYTES + 256);

    const int corr_n = NSAMP * NS * 8;
    zero_f32<<<(corr_n + 255) / 256, 256, 0, stream>>>(corr, corr_n);
    grd_norms_k<<<NSAMP, 256, 0, stream>>>(grd, norms);
    grd_swizzle_bf16<<<CCH * HK, 256, 0, stream>>>(grd, gbf);
    sat_energy_k<<<dim3(NSAMP, HK), 256, 0, stream>>>(sat, E);
    integral2d_k<<<NSAMP, 256, 0, stream>>>(E);
    corr_wmma_k<<<dim3(NSAMP, SGRP, KSPLIT), 256, 0, stream>>>(sat, gbf, corr);
    finalize_k<<<NSAMP, 512, 0, stream>>>(corr, E, norms, out, peaks);
    make_mask_k<<<dim3(NSAMP, HK), 256, 0, stream>>>(grd, norms, peaks, out + 64);
}
